// Module1KeyPruningNetwork_48842368090522
// MI455X (gfx1250) — matrix operand order repaired and checked by reading, not compiled, not run on hardware
//
#include <hip/hip_runtime.h>
#include <hip/hip_bf16.h>

#define NUM_BINS    128
#define NUM_FREQS   64
#define NUM_KERNELS 3
#define MLP_HIDDEN  64
#define NUM_KEYS    8192
#define TILE_KEYS   16
#define LOG2E       1.44269504088896340736f
#define RLOG2E      0.69314718055994530942f
#define LOG10_2     0.30102999566398119521f

typedef __attribute__((ext_vector_type(16))) _Float16 v16h;
typedef __attribute__((ext_vector_type(8)))  float    v8f;

__device__ __forceinline__ float fast_exp2(float x) {
#if __has_builtin(__builtin_amdgcn_exp2f)
  return __builtin_amdgcn_exp2f(x);
#else
  return exp2f(x);
#endif
}
__device__ __forceinline__ float fast_log2(float x) {
#if __has_builtin(__builtin_amdgcn_logf)
  return __builtin_amdgcn_logf(x);
#else
  return log2f(x);
#endif
}
__device__ __forceinline__ float fast_rcp(float x) {
#if __has_builtin(__builtin_amdgcn_rcpf)
  return __builtin_amdgcn_rcpf(x);
#else
  return 1.0f / x;
#endif
}

// ---------------------------------------------------------------------------
// Kernel 1: fold cos/sin/kappa/log2e into a (B,F,M) float4 table in scratch.
//   t = { kappa*cos(mu+ref)*log2e, kappa*sin(mu+ref)*log2e, -kappa*log2e, weight }
// so the hot loop is  w * exp2( x̂*t.x + ŷ*t.y + t.z ).
// ---------------------------------------------------------------------------
__global__ __launch_bounds__(256) void kpn_precompute(
    const float* __restrict__ mu, const float* __restrict__ kappa,
    const float* __restrict__ weight, const float* __restrict__ ref_angles,
    float4* __restrict__ table) {
  int idx = blockIdx.x * 256 + threadIdx.x;
  if (idx >= NUM_BINS * NUM_FREQS * NUM_KERNELS) return;
  int f = (idx / NUM_KERNELS) % NUM_FREQS;
  float mue = mu[idx] + ref_angles[f];
  float kap = kappa[idx];
  float s, c;
  __sincosf(mue, &s, &c);
  float4 t;
  t.x = kap * c * LOG2E;
  t.y = kap * s * LOG2E;
  t.z = -kap * LOG2E;
  t.w = weight[idx];
  table[idx] = t;
}

// ---------------------------------------------------------------------------
// Kernel 2: main. One block = 16 keys.
//   Phase 0: normalize K -> (cosA, sinA, mag) in LDS
//   Phase 1: logits[16][128] (201M v_exp_f32 total across grid)
//   Phase 2: MLP GEMM on v_wmma_f32_16x16x32_f16, ReLU, W2-dot via ds_add_f32,
//            sigmoid * position weight.
// ---------------------------------------------------------------------------
__global__ __launch_bounds__(256) void kpn_main(
    const float* __restrict__ K, const int* __restrict__ key_pos,
    const float4* __restrict__ table, const float* __restrict__ bias,
    const float* __restrict__ W1, const float* __restrict__ b1,
    const float* __restrict__ W2, const float* __restrict__ b2,
    const float* __restrict__ anchor_raw, float* __restrict__ out) {
  __shared__ float4   sXYM[TILE_KEYS][NUM_FREQS + 1];   // +1 pad: kill bank conflicts
  __shared__ _Float16 sLogit[TILE_KEYS][NUM_BINS + 8];  // +8 pad
  __shared__ float    sAcc[TILE_KEYS];

  const int tid = threadIdx.x;
  const int n0  = blockIdx.x * TILE_KEYS;

  if (tid < TILE_KEYS) sAcc[tid] = 0.0f;

  // ---------------- Phase 0: per-(key,freq) normalization --------------------
  for (int p = tid; p < TILE_KEYS * NUM_FREQS; p += 256) {
    int n = p >> 6, f = p & 63;
    float x = K[(n0 + n) * (2 * NUM_FREQS) + 2 * f + 0];
    float y = K[(n0 + n) * (2 * NUM_FREQS) + 2 * f + 1];
    float mag = __builtin_sqrtf(x * x + y * y);
    float inv = (mag > 0.0f) ? fast_rcp(mag) : 0.0f;
    float ca  = (mag > 0.0f) ? x * inv : 1.0f;  // atan2(0,0)=0 -> cos=1,sin=0
    float sa  = y * inv;
    sXYM[n][f] = make_float4(ca, sa, mag, 0.0f);
  }
  __syncthreads();

  // ---------------- Phase 1: logits ------------------------------------------
  {
    const int n  = tid & 15;   // key within tile
    const int bg = tid >> 4;   // bin group: 8 bins each
    float acc[8];
#pragma unroll
    for (int i = 0; i < 8; ++i) acc[i] = 0.0f;

    for (int f = 0; f < NUM_FREQS; ++f) {
      float4 a = sXYM[n][f];
#pragma unroll
      for (int i = 0; i < 8; ++i) {
        int b = bg * 8 + i;
        const float4* t = table + ((size_t)b * NUM_FREQS + f) * NUM_KERNELS;
        float s = 0.0f;
#pragma unroll
        for (int m = 0; m < NUM_KERNELS; ++m) {
          float4 tm = t[m];  // L2-resident broadcast table
          float e = fast_exp2(fmaf(a.x, tm.x, fmaf(a.y, tm.y, tm.z)));
          s = fmaf(tm.w, e, s);
        }
        acc[i] = fmaf(a.z, s, acc[i]);
      }
    }
#pragma unroll
    for (int i = 0; i < 8; ++i) {
      int b = bg * 8 + i;
      sLogit[n][b] = (_Float16)(acc[i] + bias[b]);
    }
  }
  __syncthreads();

  // ---------------- Phase 2: MLP via WMMA ------------------------------------
  const int wave = tid >> 5;
  const int lane = tid & 31;
  if (wave < 4) {                      // 4 waves, one 16x16 N-tile of h each
    const int tj   = wave;             // hidden-tile index
    const int rc   = lane & 15;
    const int half = lane >> 4;
    v8f d = {};
#pragma unroll
    for (int c = 0; c < 4; ++c) {      // K = 128 in chunks of 32
      v16h a, bf;
      const int kbase = half * 16;     // contiguous k = 16*half + e (matches B)
#pragma unroll
      for (int e = 0; e < 16; ++e) {
        int kv = kbase + e;
        a[e] = sLogit[rc][c * 32 + kv];
      }
      const int col = tj * 16 + rc;    // B = W1^T column (hidden unit)
      const float* wrow = W1 + (size_t)col * NUM_BINS + c * 32 + half * 16;
#pragma unroll
      for (int e = 0; e < 16; ++e) bf[e] = (_Float16)wrow[e];
      d = __builtin_amdgcn_wmma_f32_16x16x32_f16(
          /*neg_a=*/false, a, /*neg_b=*/false, bf,
          /*c_mod=*/(short)0, d, /*reuse_a=*/false, /*reuse_b=*/false);
    }
    // epilogue: ReLU(h + b1) * W2, reduce over hidden via LDS float atomics
    const int colh = tj * 16 + rc;
    float b1v = b1[colh];
    float w2v = W2[colh];
#pragma unroll
    for (int r = 0; r < 8; ++r) {
      int key = r + half * 8;          // D layout: VGPR r -> M = r + 8*half
      float h = d[r] + b1v;
      h = h > 0.0f ? h : 0.0f;
      atomicAdd(&sAcc[key], h * w2v);  // ds_add_f32
    }
  }
  __syncthreads();

  // ---------------- Finalize: position weight + sigmoid ----------------------
  if (tid < TILE_KEYS) {
    float aw[3];
#pragma unroll
    for (int i = 0; i < 3; ++i) {      // softplus(anchor_raw)
      float x  = anchor_raw[i];
      float ex = fast_exp2(x * LOG2E);
      aw[i] = fast_log2(1.0f + ex) * RLOG2E;
    }
    float pf = (float)key_pos[n0 + tid];
    if (pf < 1.0f) pf = 1.0f;
    float lp = fast_log2(pf) * LOG10_2;  // log10(position)
    float w;
    if (lp >= 5.0f)       w = aw[2];
    else if (lp >= 4.0f) { float t = lp - 4.0f; w = aw[1] * (1.0f - t) + aw[2] * t; }
    else if (lp >= 3.0f) { float t = lp - 3.0f; w = aw[0] * (1.0f - t) + aw[1] * t; }
    else                  w = aw[0];
    float m = (sAcc[tid] + b2[0]) * w;
    out[n0 + tid] = fast_rcp(1.0f + fast_exp2(-m * LOG2E));  // sigmoid
  }
}

extern "C" void kernel_launch(void* const* d_in, const int* in_sizes, int n_in,
                              void* d_out, int out_size, void* d_ws, size_t ws_size,
                              hipStream_t stream) {
  (void)in_sizes; (void)n_in; (void)out_size; (void)ws_size;
  const float* K       = (const float*)d_in[0];
  const int*   keypos  = (const int*)  d_in[1];
  const float* ref     = (const float*)d_in[2];
  const float* mu      = (const float*)d_in[3];
  const float* kappa   = (const float*)d_in[4];
  const float* weight  = (const float*)d_in[5];
  const float* bias    = (const float*)d_in[6];
  const float* W1      = (const float*)d_in[7];
  const float* b1      = (const float*)d_in[8];
  const float* W2      = (const float*)d_in[9];
  const float* b2      = (const float*)d_in[10];
  const float* anchor  = (const float*)d_in[11];
  float*  out   = (float*)d_out;
  float4* table = (float4*)d_ws;  // 24576 * 16B = 384 KB scratch

  const int tot = NUM_BINS * NUM_FREQS * NUM_KERNELS;
  kpn_precompute<<<(tot + 255) / 256, 256, 0, stream>>>(mu, kappa, weight, ref, table);
  kpn_main<<<NUM_KEYS / TILE_KEYS, 256, 0, stream>>>(K, keypos, table, bias,
                                                     W1, b1, W2, b2, anchor, out);
}
